// Pairwise_89378269430083
// MI455X (gfx1250) — compile-verified, matrix operand order checked by reading, not run on hardware
//
#include <hip/hip_runtime.h>
#include <hip/hip_bf16.h>
#include <math.h>

// Problem constants (B,N,D,H) = (4,512,256,256)
#define BB 4
#define NN 512
#define DD 256
#define HH 256
#define MROWS (BB * NN)   // 2048 rows through the MLP
#define NTILE (NN / 16)   // 32 row/col tiles per batch
#define NPAIR (NTILE * (NTILE + 1) / 2)  // 528 upper-triangle tile pairs

typedef __attribute__((ext_vector_type(16))) __bf16 v16bf;
typedef __attribute__((ext_vector_type(8)))  float  v8f;

// Native CDNA5 transcendental tanh (V_TANH_F32): single trans-pipe op instead of
// the branchy libm expansion. Builtin when available; raw asm fallback with a
// v_nop to satisfy the "1 independent op after TRANS" hazard rule (ISA 7.4).
__device__ __forceinline__ float fast_tanh(float x) {
#if __has_builtin(__builtin_amdgcn_tanhf)
    return __builtin_amdgcn_tanhf(x);
#else
    float r;
    asm volatile("v_tanh_f32 %0, %1\n\tv_nop" : "=v"(r) : "v"(x));
    return r;
#endif
}

__device__ __forceinline__ unsigned short f32_to_bf16(float x) {
    unsigned int b = __float_as_uint(x);
    unsigned int r = b + 0x7FFFu + ((b >> 16) & 1u);   // round-to-nearest-even
    return (unsigned short)(r >> 16);
}

__global__ void cvt_bf16_kernel(const float* __restrict__ in,
                                unsigned short* __restrict__ out, int n) {
    int i = blockIdx.x * blockDim.x + threadIdx.x;
    if (i < n) out[i] = f32_to_bf16(in[i]);
}

// C[M,256] = act(A[M,256] (bf16, row-major) x W[256,256]^T (bf16, row-major) + bias)
// One wave32 owns one 16x16 output tile; K = 256/32 = 8 WMMAs, fully unrolled.
__global__ __launch_bounds__(256) void gemm_bf16_wmma_kernel(
    const unsigned short* __restrict__ A,
    const unsigned short* __restrict__ W,
    const float* __restrict__ bias,
    float* __restrict__ outF,            // optional f32 result (or nullptr)
    unsigned short* __restrict__ outB,   // optional bf16 result (or nullptr)
    int do_tanh)
{
    const int lane = threadIdx.x & 31;
    const int wave = threadIdx.x >> 5;
    const int tile = blockIdx.x * 8 + wave;          // 2048 tiles total
    const int tiles_n = HH / 16;                     // 16
    const int m0 = (tile / tiles_n) * 16;
    const int n0 = (tile % tiles_n) * 16;

    const int rowA = m0 + (lane & 15);
    const int rowW = n0 + (lane & 15);
    const int ksel = (lane >> 4) * 8;

    v8f acc = {};
#pragma unroll
    for (int kt = 0; kt < DD; kt += 32) {
        const int kb = kt + ksel;
        union { v16bf v; uint4 q[2]; } a, b;
        a.q[0] = *(const uint4*)(A + rowA * DD + kb);
        a.q[1] = *(const uint4*)(A + rowA * DD + kb + 16);
        b.q[0] = *(const uint4*)(W + rowW * DD + kb);
        b.q[1] = *(const uint4*)(W + rowW * DD + kb + 16);
        acc = __builtin_amdgcn_wmma_f32_16x16x32_bf16(
            /*neg_a=*/false, a.v, /*neg_b=*/false, b.v,
            /*c_mod=*/(short)0, acc, /*reuse_a=*/false, /*reuse_b=*/false);
    }

    // C/D layout: VGPR r -> row = m0 + r + 8*(lane>=16), col = n0 + (lane&15)
    const int col   = n0 + (lane & 15);
    const float bv  = bias[col];
    const int rbase = m0 + ((lane >> 4) * 8);
#pragma unroll
    for (int r = 0; r < 8; ++r) {
        const int row = rbase + r;
        float val = acc[r] + bv;
        if (do_tanh) val = fast_tanh(val);
        if (outF) outF[row * HH + col] = val;
        if (outB) outB[row * HH + col] = f32_to_bf16(val);
    }
}

// Fused pairwise + symmetrize:
//   out[b,i,j] = out[b,j,i] = 0.5 * ( v.tanh(Q_i + K_j) + v.tanh(Q_j + K_i) )
// One 256-thread block per upper-triangle tile pair (ti <= tj). Four 16x256 f32
// tiles + v staged in LDS (65 KB of 320 KB). Each Q/K row reused 16x from LDS,
// so the loop is bound by the trans pipe (v_tanh_f32), not HBM.
__global__ __launch_bounds__(256) void pairwise_sym_kernel(
    const float* __restrict__ Q, const float* __restrict__ K,
    const float* __restrict__ v, float* __restrict__ out)
{
    __shared__ float QsI[16 * HH];
    __shared__ float KsJ[16 * HH];
    __shared__ float QsJ[16 * HH];
    __shared__ float KsI[16 * HH];
    __shared__ float Vs[HH];

    const int blk = blockIdx.x;            // BB * NPAIR blocks
    const int b   = blk / NPAIR;
    int p = blk % NPAIR;
    int ti = 0, len = NTILE;               // map p -> (ti, tj), ti <= tj
    while (p >= len) { p -= len; ++ti; --len; }
    const int tj = ti + p;
    const int i0 = ti * 16, j0 = tj * 16;
    const int t  = threadIdx.x;

    // Each 16x256 tile is a contiguous 4096-float span -> straight float4 copies.
    {
        const float4* Qi = (const float4*)(Q + (size_t)(b * NN + i0) * HH);
        const float4* Qj = (const float4*)(Q + (size_t)(b * NN + j0) * HH);
        const float4* Ki = (const float4*)(K + (size_t)(b * NN + i0) * HH);
        const float4* Kj = (const float4*)(K + (size_t)(b * NN + j0) * HH);
        float4* qI = (float4*)QsI; float4* qJ = (float4*)QsJ;
        float4* kI = (float4*)KsI; float4* kJ = (float4*)KsJ;
        for (int idx = t; idx < 16 * HH / 4; idx += 256) {
            qI[idx] = Qi[idx];
            qJ[idx] = Qj[idx];
            kI[idx] = Ki[idx];
            kJ[idx] = Kj[idx];
        }
        if (t < HH / 4) ((float4*)Vs)[t] = ((const float4*)v)[t];
    }
    __syncthreads();

    const int il = t >> 4, jl = t & 15;
    const float4* qi = (const float4*)(QsI + il * HH);
    const float4* kj = (const float4*)(KsJ + jl * HH);
    const float4* qj = (const float4*)(QsJ + jl * HH);
    const float4* ki = (const float4*)(KsI + il * HH);
    const float4* vr = (const float4*)Vs;

    float s1 = 0.0f, s2 = 0.0f;
#pragma unroll 2
    for (int h = 0; h < HH / 4; ++h) {
        const float4 a = qi[h], c = kj[h], d = qj[h], e = ki[h], w = vr[h];
        s1 += w.x * fast_tanh(a.x + c.x);
        s1 += w.y * fast_tanh(a.y + c.y);
        s1 += w.z * fast_tanh(a.z + c.z);
        s1 += w.w * fast_tanh(a.w + c.w);
        s2 += w.x * fast_tanh(d.x + e.x);
        s2 += w.y * fast_tanh(d.y + e.y);
        s2 += w.z * fast_tanh(d.z + e.z);
        s2 += w.w * fast_tanh(d.w + e.w);
    }
    const float r = 0.5f * (s1 + s2);
    out[((size_t)b * NN + i0 + il) * NN + j0 + jl] = r;
    out[((size_t)b * NN + j0 + jl) * NN + i0 + il] = r;
}

// ---------------- workspace layout (bytes) ----------------
#define OFF_XB   (size_t)0                      // X  bf16   1 MB
#define OFF_W1B  (OFF_XB  + (size_t)MROWS*DD*2)
#define OFF_W2B  (OFF_W1B + (size_t)HH*DD*2)
#define OFF_WQB  (OFF_W2B + (size_t)HH*HH*2)
#define OFF_WKB  (OFF_WQB + (size_t)HH*HH*2)
#define OFF_H1B  (OFF_WKB + (size_t)HH*HH*2)    // h1 bf16   1 MB
#define OFF_H2B  (OFF_H1B + (size_t)MROWS*HH*2) // h2 bf16   1 MB
#define OFF_QF   (OFF_H2B + (size_t)MROWS*HH*2) // Q  f32    2 MB
#define OFF_KF   (OFF_QF  + (size_t)MROWS*HH*4) // K  f32    2 MB

extern "C" void kernel_launch(void* const* d_in, const int* in_sizes, int n_in,
                              void* d_out, int out_size, void* d_ws, size_t ws_size,
                              hipStream_t stream) {
    (void)in_sizes; (void)n_in; (void)out_size; (void)ws_size;
    const float* X  = (const float*)d_in[0];
    // d_in[1] = Y (unused, as in the reference)
    const float* W1 = (const float*)d_in[2];
    const float* b1 = (const float*)d_in[3];
    const float* W2 = (const float*)d_in[4];
    const float* b2 = (const float*)d_in[5];
    const float* Wq = (const float*)d_in[6];
    const float* bq = (const float*)d_in[7];
    const float* Wk = (const float*)d_in[8];
    const float* bk = (const float*)d_in[9];
    const float* v  = (const float*)d_in[10];
    // d_in[11] = k (unused)

    char* ws = (char*)d_ws;
    unsigned short* Xb  = (unsigned short*)(ws + OFF_XB);
    unsigned short* W1b = (unsigned short*)(ws + OFF_W1B);
    unsigned short* W2b = (unsigned short*)(ws + OFF_W2B);
    unsigned short* Wqb = (unsigned short*)(ws + OFF_WQB);
    unsigned short* Wkb = (unsigned short*)(ws + OFF_WKB);
    unsigned short* h1b = (unsigned short*)(ws + OFF_H1B);
    unsigned short* h2b = (unsigned short*)(ws + OFF_H2B);
    float*          Qf  = (float*)(ws + OFF_QF);
    float*          Kf  = (float*)(ws + OFF_KF);
    float*          out = (float*)d_out;

    // 1) convert f32 -> bf16 for WMMA operands
    cvt_bf16_kernel<<<(MROWS * DD + 255) / 256, 256, 0, stream>>>(X,  Xb,  MROWS * DD);
    cvt_bf16_kernel<<<(HH * DD + 255) / 256, 256, 0, stream>>>(W1, W1b, HH * DD);
    cvt_bf16_kernel<<<(HH * HH + 255) / 256, 256, 0, stream>>>(W2, W2b, HH * HH);
    cvt_bf16_kernel<<<(HH * HH + 255) / 256, 256, 0, stream>>>(Wq, Wqb, HH * HH);
    cvt_bf16_kernel<<<(HH * HH + 255) / 256, 256, 0, stream>>>(Wk, Wkb, HH * HH);

    // 2) WMMA GEMM chain: 2048 tiles of 16x16, 8 waves/block -> 256 blocks each
    const int gemm_blocks = (MROWS / 16) * (HH / 16) / 8;  // 256
    gemm_bf16_wmma_kernel<<<gemm_blocks, 256, 0, stream>>>(Xb,  W1b, b1, nullptr, h1b, /*tanh=*/1);
    gemm_bf16_wmma_kernel<<<gemm_blocks, 256, 0, stream>>>(h1b, W2b, b2, nullptr, h2b, /*tanh=*/0);
    gemm_bf16_wmma_kernel<<<gemm_blocks, 256, 0, stream>>>(h2b, Wqb, bq, Qf, nullptr, /*tanh=*/0);
    gemm_bf16_wmma_kernel<<<gemm_blocks, 256, 0, stream>>>(h2b, Wkb, bk, Kf, nullptr, /*tanh=*/0);

    // 3) fused additive-attention scores + symmetrize (trans-pipe bound)
    pairwise_sym_kernel<<<BB * NPAIR, 256, 0, stream>>>(Qf, Kf, v, out);
}
